// SessionSimilarityAggregation_24077586661600
// MI455X (gfx1250) — matrix operand order repaired and checked by reading, not compiled, not run on hardware
//
#include <hip/hip_runtime.h>
#include <hip/hip_fp16.h>

// SessionSimilarityAggregation for MI455X (gfx1250, wave32, WMMA).
// Kernel 1: L2-normalize rows of sess_emb [16384,256] f32 -> f16 in d_ws.
// Kernel 2: fused cos-sim GEMM (v_wmma_f32_16x16x32_f16, K=256) with a
//           4-deep async global->LDS ring (global_load_async_to_lds_b128,
//           s_wait_asynccnt, 1 barrier/tile), batched ds_load_b128 fragment
//           fetch pipelined against the WMMA chain, streaming per-row top-3
//           + softmax + gather-sum -> d_out.

typedef __attribute__((ext_vector_type(16))) _Float16 v16h;
typedef __attribute__((ext_vector_type(8)))  _Float16 v8h;
typedef __attribute__((ext_vector_type(8)))  float    v8f;
typedef __attribute__((ext_vector_type(4)))  int      v4i;

typedef __attribute__((address_space(1))) v4i gv4i;  // global int4 (b128 payload)
typedef __attribute__((address_space(3))) v4i lv4i;  // LDS int4

#define NROWS    16384
#define DIM      256
#define NTILES   (NROWS / 16)
#define LDSPITCH 264   // 256 f16 + 8 f16 pad: lanes land on distinct bank groups
#define NBUF     4     // async ring depth (3 tiles of copy lead time)

// ---------------------------------------------------------------------------
// Kernel 1: one wave per row; butterfly-reduce sum of squares, scale, f16 out.
// ---------------------------------------------------------------------------
__global__ __launch_bounds__(256) void normalize_rows_f16(
    const float* __restrict__ x, _Float16* __restrict__ nrm) {
  const int wave = threadIdx.x >> 5;
  const int lane = threadIdx.x & 31;
  const int row  = blockIdx.x * 8 + wave;

  const float* rp = x + (size_t)row * DIM + lane * 8;
  float4 f0 = *(const float4*)(rp);
  float4 f1 = *(const float4*)(rp + 4);

  float s = f0.x * f0.x + f0.y * f0.y + f0.z * f0.z + f0.w * f0.w +
            f1.x * f1.x + f1.y * f1.y + f1.z * f1.z + f1.w * f1.w;
  #pragma unroll
  for (int off = 16; off >= 1; off >>= 1) s += __shfl_xor(s, off, 32);

  const float scale = 1.0f / fmaxf(sqrtf(s), 1e-12f);

  v8h h;
  h[0] = (_Float16)(f0.x * scale); h[1] = (_Float16)(f0.y * scale);
  h[2] = (_Float16)(f0.z * scale); h[3] = (_Float16)(f0.w * scale);
  h[4] = (_Float16)(f1.x * scale); h[5] = (_Float16)(f1.y * scale);
  h[6] = (_Float16)(f1.z * scale); h[7] = (_Float16)(f1.w * scale);
  *(v8h*)(nrm + (size_t)row * DIM + lane * 8) = h;
}

// ---------------------------------------------------------------------------
// Async-copy one 16x256 f16 B tile (8 KB, contiguous rows) into padded LDS.
// Each of 256 threads moves 2 x 16B chunks via ASYNCcnt-tracked DMA.
// ---------------------------------------------------------------------------
__device__ __forceinline__ void async_copy_tile(const _Float16* __restrict__ nrm,
                                                _Float16* lds_base, int tile,
                                                int tid) {
  const _Float16* g = nrm + (size_t)tile * (16 * DIM);
  #pragma unroll
  for (int half = 0; half < 2; ++half) {
    const int c   = tid + half * 256;       // b128 chunk id, 512 per tile
    const int r   = c >> 5;                 // tile row (32 chunks per row)
    const int off = (c & 31) * 8;           // f16 offset within row
    const _Float16* gp = g + c * 8;
    _Float16*       lp = lds_base + r * LDSPITCH + off;
    __builtin_amdgcn_global_load_async_to_lds_b128((gv4i*)gp, (lv4i*)lp, 0, 0);
  }
}

// ---------------------------------------------------------------------------
// Kernel 2: each wave owns 16 rows; A fragments (16x256 f16) resident in
// VGPRs; B tiles staged through a 4-deep LDS ring filled by async DMA.
// ---------------------------------------------------------------------------
__global__ __launch_bounds__(256) void cosine_topk_agg(
    const _Float16* __restrict__ nrm, const float* __restrict__ x,
    float* __restrict__ out) {
  __shared__ alignas(64) _Float16 btile[NBUF][16 * LDSPITCH];

  const int tid     = threadIdx.x;
  const int wave    = tid >> 5;
  const int lane    = tid & 31;
  const int rowBase = blockIdx.x * 128 + wave * 16;
  const int m       = lane & 15;   // A row / B col / C column index
  const int hi      = lane >> 4;   // K-half selector per ISA VGPR layouts

  // ---- A fragments: lane holds row (rowBase+m); K sub-chunks by lane-half.
  v16h afrag[8];
  {
    const _Float16* ar = nrm + (size_t)(rowBase + m) * DIM;
    #pragma unroll
    for (int kb = 0; kb < 8; ++kb) {
      const _Float16* p = ar + kb * 32;
      v8h lo  = *(const v8h*)(p + hi * 8);
      v8h hi8 = *(const v8h*)(p + 16 + hi * 8);
      v16h a;
      #pragma unroll
      for (int i = 0; i < 8; ++i) { a[i] = lo[i]; a[i + 8] = hi8[i]; }
      afrag[kb] = a;
    }
  }

  // Running top-3 (value,index) per C row r, replicated across the 16-lane
  // half that owns row (rowBase + 8*hi + r) per the C-matrix VGPR layout.
  float t0v[8], t1v[8], t2v[8];
  int   t0i[8], t1i[8], t2i[8];
  #pragma unroll
  for (int r = 0; r < 8; ++r) {
    t0v[r] = t1v[r] = t2v[r] = -__builtin_inff();
    t0i[r] = t1i[r] = t2i[r] = 0;
  }

  // ---- prologue: prime the ring three tiles deep (6 async loads in flight).
  async_copy_tile(nrm, &btile[0][0], 0, tid);
  async_copy_tile(nrm, &btile[1][0], 1, tid);
  async_copy_tile(nrm, &btile[2][0], 2, tid);

  for (int t = 0; t < NTILES; ++t) {
    const int buf     = t & (NBUF - 1);
    const int colBase = t * 16;

    // Tile t complete <=> at most tiles t+1,t+2 (2 loads each) outstanding.
    if (t + 2 < NTILES)      asm volatile("s_wait_asynccnt 0x4" ::: "memory");
    else if (t + 1 < NTILES) asm volatile("s_wait_asynccnt 0x2" ::: "memory");
    else                     asm volatile("s_wait_asynccnt 0x0" ::: "memory");
    __syncthreads();  // tile t resident for all waves; all done with tile t-1

    // Refill the slot that held tile t-1 (provably drained after the barrier);
    // DMA overlaps the compute below with 3 tiles of lead time.
    if (t + 3 < NTILES)
      async_copy_tile(nrm, &btile[(t + 3) & (NBUF - 1)][0], t + 3, tid);

    // ---- batch-fetch ALL B fragments into 8 live v16h registers, then run
    //      the WMMA chain. The sched_barrier pins the 16 ds_loads ahead of
    //      the WMMAs so DS latency pipelines under the matrix pipe.
    const _Float16* lb = &btile[buf][0] + m * LDSPITCH + hi * 8;
    v16h bfr[8];
    #pragma unroll
    for (int kb = 0; kb < 8; ++kb) {
      v8h b0 = *(const v8h*)(lb + kb * 32);
      v8h b1 = *(const v8h*)(lb + kb * 32 + 16);
      v16h b;
      #pragma unroll
      for (int i = 0; i < 8; ++i) { b[i] = b0[i]; b[i + 8] = b1[i]; }
      bfr[kb] = b;
    }
    __builtin_amdgcn_sched_barrier(0);  // keep loads above, WMMAs below

    v8f c = {0.f, 0.f, 0.f, 0.f, 0.f, 0.f, 0.f, 0.f};
    #pragma unroll
    for (int kb = 0; kb < 8; ++kb) {
      c = __builtin_amdgcn_wmma_f32_16x16x32_f16(
              /*neg_a=*/false, afrag[kb], /*neg_b=*/false, bfr[kb],
              /*c_mod=*/(short)0, c, /*reuse_a=*/false, /*reuse_b=*/false);
    }

    // ---- cheap rejection: only extract when some lane beats the 3rd-best.
    bool cand = false;
    #pragma unroll
    for (int r = 0; r < 8; ++r) cand |= (c[r] > t2v[r]);
    if (__builtin_amdgcn_ballot_w32(cand)) {
      #pragma unroll
      for (int r = 0; r < 8; ++r) {
        float v  = c[r];
        int   ci = colBase + m;
        #pragma unroll
        for (int round = 0; round < 3; ++round) {
          float mv = v; int mi = ci;
          #pragma unroll
          for (int off = 8; off >= 1; off >>= 1) {  // stays within 16-lane half
            float ov = __shfl_xor(mv, off, 32);
            int   oi = __shfl_xor(mi, off, 32);
            if (ov > mv || (ov == mv && oi < mi)) { mv = ov; mi = oi; }
          }
          if (mv > t0v[r]) {
            t2v[r] = t1v[r]; t2i[r] = t1i[r];
            t1v[r] = t0v[r]; t1i[r] = t0i[r];
            t0v[r] = mv;     t0i[r] = mi;
          } else if (mv > t1v[r]) {
            t2v[r] = t1v[r]; t2i[r] = t1i[r];
            t1v[r] = mv;     t1i[r] = mi;
          } else if (mv > t2v[r]) {
            t2v[r] = mv;     t2i[r] = mi;
          }
          if (ci == mi) v = -__builtin_inff();  // drop winner for next round
        }
      }
    }
  }

  // ---- epilogue: softmax over the 3 sims, weighted sum of raw f32 rows.
  // Half-wave hi owns rows rowBase + 8*hi + r; its 16 lanes split 256 cols.
  #pragma unroll
  for (int r = 0; r < 8; ++r) {
    const int row = rowBase + 8 * hi + r;
    float w0 = 1.0f;
    float w1 = __expf(t1v[r] - t0v[r]);
    float w2 = __expf(t2v[r] - t0v[r]);
    const float inv = 1.0f / (w0 + w1 + w2);
    w0 *= inv; w1 *= inv; w2 *= inv;

    const float* s0 = x + (size_t)t0i[r] * DIM;
    const float* s1 = x + (size_t)t1i[r] * DIM;
    const float* s2 = x + (size_t)t2i[r] * DIM;
    float* op = out + (size_t)row * DIM;

    const int c0 = m * 16;
    #pragma unroll
    for (int q = 0; q < 4; ++q) {
      const int cc = c0 + q * 4;
      float4 a = *(const float4*)(s0 + cc);
      float4 b = *(const float4*)(s1 + cc);
      float4 d = *(const float4*)(s2 + cc);
      float4 o;
      o.x = w0 * a.x + w1 * b.x + w2 * d.x;
      o.y = w0 * a.y + w1 * b.y + w2 * d.y;
      o.z = w0 * a.z + w1 * b.z + w2 * d.z;
      o.w = w0 * a.w + w1 * b.w + w2 * d.w;
      *(float4*)(op + cc) = o;
    }
  }
}

// ---------------------------------------------------------------------------
extern "C" void kernel_launch(void* const* d_in, const int* in_sizes, int n_in,
                              void* d_out, int out_size, void* d_ws,
                              size_t ws_size, hipStream_t stream) {
  (void)in_sizes; (void)n_in; (void)out_size; (void)ws_size;
  const float* x   = (const float*)d_in[0];
  float*       out = (float*)d_out;
  _Float16*    nrm = (_Float16*)d_ws;  // 16384*256 f16 = 8 MB scratch

  normalize_rows_f16<<<NROWS / 8, 256, 0, stream>>>(x, nrm);
  cosine_topk_agg<<<NROWS / 128, 256, 0, stream>>>(nrm, x, out);
}
